// CaMoE_System_61546881352173
// MI455X (gfx1250) — compile-verified, matrix operand order checked by reading, not run on hardware
//
#include <hip/hip_runtime.h>
#include <hip/hip_bf16.h>

// ---------------------------------------------------------------------------
// CaMoE block: out = x + sum_top2 coef * relu(ln(x) @ W1[e])^2 @ W2[e]
// B=4 T=2048 C=1024 E=8 H=4096.
// LN->bf16 | route/compact | pack W fp32->bf16 fragment-major | grouped GEMM:
// M=32 token tiles, 512-thread (16-wave) blocks, v_wmma_f32_16x16x32_bf16,
// async-to-LDS staging of A tiles, each B fragment feeds 2 WMMAs.
// (M=64 variant spilled to scratch; M=32 is the no-spill sweet spot.)
// ---------------------------------------------------------------------------

#define Bb 4
#define Tt 2048
#define Cc 1024
#define Ee 8
#define Hh 4096
#define NTOK   (Bb * Tt)       // 8192 tokens
#define NSLOT  (NTOK * 2)      // 16384 routed slots
#define TM     32              // tokens per tile
#define MAXTIL (NSLOT / TM)    // 512 worst-case tiles per expert
#define NCHUNK 256             // hidden cols per chunk (16 waves * 16)

typedef __attribute__((ext_vector_type(16))) __bf16 v16bf;
typedef __attribute__((ext_vector_type(8)))  float  v8f;

// K-pair base for 16-bit A/B fragments (ISA 7.12.2, 16-bit A 16x32 table):
// VGPR v in [0,3]: K = 2v (+8 for lanes 16-31); v in [4,7]: K = 16+2(v-4) (+8)
__device__ __forceinline__ int kbase(int v, int half) {
    return ((v & 4) << 2) | ((v & 3) << 1) | (half << 3);
}

// ------------------------------- LayerNorm ---------------------------------
__global__ void ln_kernel(const float* __restrict__ x,
                          const float* __restrict__ gamma,
                          const float* __restrict__ beta,
                          __bf16* __restrict__ h) {
    __shared__ float s_sum[256];
    __shared__ float s_sq[256];
    __shared__ float s_stats[2];
    const int row = blockIdx.x;
    const int tid = threadIdx.x;
    const float4 v = ((const float4*)(x + (size_t)row * Cc))[tid];
    float sum = v.x + v.y + v.z + v.w;
    float sq  = v.x * v.x + v.y * v.y + v.z * v.z + v.w * v.w;
    s_sum[tid] = sum; s_sq[tid] = sq;
    __syncthreads();
    for (int s = 128; s > 0; s >>= 1) {
        if (tid < s) { s_sum[tid] += s_sum[tid + s]; s_sq[tid] += s_sq[tid + s]; }
        __syncthreads();
    }
    if (tid == 0) {
        float mu  = s_sum[0] * (1.0f / Cc);
        float var = s_sq[0] * (1.0f / Cc) - mu * mu;
        s_stats[0] = mu;
        s_stats[1] = rsqrtf(var + 1e-5f);
    }
    __syncthreads();
    const float mu = s_stats[0], rs = s_stats[1];
    const float4 g = ((const float4*)gamma)[tid];
    const float4 b = ((const float4*)beta)[tid];
    __bf16* hr = h + (size_t)row * Cc + tid * 4;
    hr[0] = (__bf16)((v.x - mu) * rs * g.x + b.x);
    hr[1] = (__bf16)((v.y - mu) * rs * g.y + b.y);
    hr[2] = (__bf16)((v.z - mu) * rs * g.z + b.z);
    hr[3] = (__bf16)((v.w - mu) * rs * g.w + b.w);
}

// -------------------------------- Routing ----------------------------------
__global__ void zero_kernel(int* __restrict__ counts) {
    if (threadIdx.x < 2 * Ee) counts[threadIdx.x] = 0;
}

__global__ void count_kernel(const int* __restrict__ winners, int* __restrict__ counts) {
    int i = blockIdx.x * blockDim.x + threadIdx.x;
    if (i < NSLOT) atomicAdd(&counts[winners[i]], 1);
}

__global__ void scan_kernel(const int* __restrict__ counts, int* __restrict__ offsets) {
    if (threadIdx.x == 0 && blockIdx.x == 0) {
        int acc = 0;
        for (int e = 0; e < Ee; ++e) { offsets[e] = acc; acc += counts[e]; }
        offsets[Ee] = acc;
    }
}

__global__ void fill_kernel(const int* __restrict__ winners,
                            const float* __restrict__ weights,
                            const int* __restrict__ offsets,
                            int* __restrict__ counts2,
                            int* __restrict__ tok_ids,
                            float* __restrict__ coefs) {
    int i = blockIdx.x * blockDim.x + threadIdx.x;
    if (i < NSLOT) {
        int e = winners[i];
        int pos = atomicAdd(&counts2[e], 1);
        int slot = offsets[e] + pos;
        tok_ids[slot] = i >> 1;        // token index
        coefs[slot]   = weights[i];    // its combine weight
    }
}

// -------------------- Weight repack: fp32 -> bf16 fragments -----------------
// W: [E][K][N] fp32 row-major. P: tiles of (32k x 16n), fragment-major:
// P[((e*(N/16)+nb)*(K/32)+kt)*512 + lane*16 + j]  with
// j = 2v+p  ->  W[e][kt*32 + kbase(v, lane>>4) + p][nb*16 + (lane&15)]
// So the GEMM's per-lane B fragment is one contiguous, 32B-aligned load.
__global__ void pack_kernel(const float* __restrict__ W, __bf16* __restrict__ P,
                            int K, int N) {
    size_t t = (size_t)blockIdx.x * blockDim.x + threadIdx.x;
    int lane = (int)(t & 31);
    size_t tile = t >> 5;                      // ((e*(N/16)+nb)*(K/32)+kt)
    int kt = (int)(tile % (size_t)(K / 32));
    size_t en = tile / (size_t)(K / 32);
    int nb = (int)(en % (size_t)(N / 16));
    int e  = (int)(en / (size_t)(N / 16));
    const float* We = W + (size_t)e * K * N;
    const int half = lane >> 4, ln = lane & 15;
    const int n = nb * 16 + ln;
    __bf16* dst = P + tile * 512 + (size_t)lane * 16;
    #pragma unroll
    for (int v = 0; v < 8; ++v) {
        int k = kt * 32 + kbase(v, half);
        dst[2 * v]     = (__bf16)We[(size_t)k * N + n];
        dst[2 * v + 1] = (__bf16)We[(size_t)(k + 1) * N + n];
    }
}

// ------------------------- Fused MoE expert GEMM ---------------------------
// Block = (expert, tile of 32 routed slots). 512 threads = 16 waves.
// GEMM1: hid = relu(h @ W1[e])^2, chunks of 256 cols (bf16 in LDS)
// GEMM2: out += coef * hid @ W2[e]  (f32 accumulators persistent in VGPRs)
// Each wave: 16 hidden cols in GEMM1; 64 output cols (4 tiles x 2 M-halves)
// in GEMM2. Every B fragment is consumed by two WMMAs (M=32).
__launch_bounds__(512, 1)
__global__ void moe_kernel(const __bf16* __restrict__ h,
                           const __bf16* __restrict__ W1p,
                           const __bf16* __restrict__ W2p,
                           const int*   __restrict__ offsets,
                           const int*   __restrict__ tok_ids,
                           const float* __restrict__ coefs,
                           float* __restrict__ out) {
    extern __shared__ __bf16 smem[];
    __bf16* a_tile = smem;             // TM x Cc      = 64 KB
    __bf16* hid    = smem + TM * Cc;   // TM x NCHUNK  = 16 KB
    __shared__ int    s_tok[TM];
    __shared__ float  s_coef[TM];

    const int e     = blockIdx.y;
    const int off   = offsets[e];
    const int count = offsets[e + 1] - off;
    const int tile  = blockIdx.x;
    if (tile * TM >= count) return;

    const int tid = threadIdx.x;
    if (tid < TM) {
        int s = tile * TM + tid;
        if (s < count) { s_tok[tid] = tok_ids[off + s]; s_coef[tid] = coefs[off + s]; }
        else           { s_tok[tid] = -1;               s_coef[tid] = 0.0f; }
    }
    __syncthreads();

    { // async-stage TM h-rows into LDS: ASYNCcnt path (CDNA5 ISA)
      // padded rows clamp to token 0; their output is discarded via tok<0.
        int r = tid >> 4, seg = tid & 15;          // 32 rows x 16 segs of 128B
        int tok = s_tok[r]; if (tok < 0) tok = 0;
        unsigned long long gaddr =
            (unsigned long long)(size_t)(h + (size_t)tok * Cc + seg * 64);
        unsigned lds = (unsigned)(size_t)(void*)(a_tile + r * Cc + seg * 64);
        #pragma unroll
        for (int i = 0; i < 8; ++i) {   // 8 x 16B = 128B per thread
            asm volatile("global_load_async_to_lds_b128 %0, %1, off"
                         :: "v"(lds), "v"(gaddr) : "memory");
            lds += 16; gaddr += 16;
        }
        asm volatile("s_wait_asynccnt 0x0" ::: "memory");
    }
    __syncthreads();

    const int wave = tid >> 5;     // 0..15
    const int lane = tid & 31;
    const int half = lane >> 4;
    const int ln   = lane & 15;

    v8f out_acc[4][2];
    #pragma unroll
    for (int j = 0; j < 4; ++j) {
        out_acc[j][0] = (v8f){0, 0, 0, 0, 0, 0, 0, 0};
        out_acc[j][1] = (v8f){0, 0, 0, 0, 0, 0, 0, 0};
    }

    for (int hc = 0; hc < Hh / NCHUNK; ++hc) {
        // ---- GEMM1: wave computes hidden cols [hc*256 + wave*16, +16),
        //      rows 0..31 via two A fragments sharing each B fragment.
        const __bf16* b1 = W1p
            + (((size_t)e * (Hh / 16) + (size_t)(hc * 16 + wave)) * (Cc / 32)) * 512
            + (size_t)lane * 16;
        v8f acc0 = (v8f){0, 0, 0, 0, 0, 0, 0, 0};
        v8f acc1 = (v8f){0, 0, 0, 0, 0, 0, 0, 0};
        for (int k0 = 0; k0 < Cc; k0 += 32) {
            v16bf af0, af1;
            const __bf16* arow0 = a_tile + ln * Cc + k0;          // M = 0..15
            const __bf16* arow1 = a_tile + (ln + 16) * Cc + k0;   // M = 16..31
            #pragma unroll
            for (int v = 0; v < 8; ++v) {
                int kb = kbase(v, half);
                af0[2 * v]     = arow0[kb];
                af0[2 * v + 1] = arow0[kb + 1];
                af1[2 * v]     = arow1[kb];
                af1[2 * v + 1] = arow1[kb + 1];
            }
            v16bf bf = *(const v16bf*)b1;                    // packed fragment
            { // WGP-scope prefetch of next K tile (pulls into near caches)
                unsigned long long pre = (unsigned long long)(size_t)(b1 + 512);
                asm volatile("global_prefetch_b8 %0, off" :: "v"(pre) : "memory");
            }
            b1 += 512;
            acc0 = __builtin_amdgcn_wmma_f32_16x16x32_bf16(
                       false, af0, false, bf, (short)0, acc0, false, false);
            acc1 = __builtin_amdgcn_wmma_f32_16x16x32_bf16(
                       false, af1, false, bf, (short)0, acc1, false, false);
        }
        __syncthreads();  // previous chunk of hid fully consumed
        #pragma unroll
        for (int v = 0; v < 8; ++v) {  // D: M = v + 8*half (+16), N = lane&15
            int m = v + half * 8;
            float r0 = acc0[v]; r0 = r0 > 0.0f ? r0 : 0.0f;
            float r1 = acc1[v]; r1 = r1 > 0.0f ? r1 : 0.0f;
            hid[m * NCHUNK + wave * 16 + ln]        = (__bf16)(r0 * r0);
            hid[(m + 16) * NCHUNK + wave * 16 + ln] = (__bf16)(r1 * r1);
        }
        __syncthreads();

        // ---- GEMM2: wave owns out cols [wave*64, +64); K-chunk = 256
        #pragma unroll
        for (int j = 0; j < 4; ++j) {
            const __bf16* b2 = W2p
                + (((size_t)e * (Cc / 16) + (size_t)(wave * 4 + j)) * (Hh / 32)
                   + (size_t)(hc * 8)) * 512
                + (size_t)lane * 16;
            #pragma unroll
            for (int kk = 0; kk < 8; ++kk) {
                v16bf af0, af1;
                const __bf16* hrow0 = hid + ln * NCHUNK + kk * 32;
                const __bf16* hrow1 = hid + (ln + 16) * NCHUNK + kk * 32;
                #pragma unroll
                for (int v = 0; v < 8; ++v) {
                    int kb = kbase(v, half);
                    af0[2 * v]     = hrow0[kb];
                    af0[2 * v + 1] = hrow0[kb + 1];
                    af1[2 * v]     = hrow1[kb];
                    af1[2 * v + 1] = hrow1[kb + 1];
                }
                v16bf bf2 = *(const v16bf*)(b2 + (size_t)kk * 512);
                out_acc[j][0] = __builtin_amdgcn_wmma_f32_16x16x32_bf16(
                                    false, af0, false, bf2, (short)0, out_acc[j][0], false, false);
                out_acc[j][1] = __builtin_amdgcn_wmma_f32_16x16x32_bf16(
                                    false, af1, false, bf2, (short)0, out_acc[j][1], false, false);
            }
        }
    }

    // ---- scatter: out[tok, col] += coef[tok] * acc
    #pragma unroll
    for (int j = 0; j < 4; ++j) {
        const int col = wave * 64 + j * 16 + ln;
        #pragma unroll
        for (int part = 0; part < 2; ++part) {
            #pragma unroll
            for (int v = 0; v < 8; ++v) {
                int m = v + half * 8 + part * 16;
                int tok = s_tok[m];
                if (tok >= 0)
                    atomicAdd(out + (size_t)tok * Cc + col,
                              out_acc[j][part][v] * s_coef[m]);
            }
        }
    }
}

// ------------------------------- Launcher ----------------------------------
extern "C" void kernel_launch(void* const* d_in, const int* in_sizes, int n_in,
                              void* d_out, int out_size, void* d_ws, size_t ws_size,
                              hipStream_t stream) {
    const float* x       = (const float*)d_in[0];
    const float* weights = (const float*)d_in[1];
    const float* gamma   = (const float*)d_in[2];
    const float* beta    = (const float*)d_in[3];
    const float* W1      = (const float*)d_in[4];
    const float* W2      = (const float*)d_in[5];
    const int*   winners = (const int*)d_in[6];
    float* out = (float*)d_out;

    // workspace carve-up
    char* ws = (char*)d_ws;
    size_t o = 0;
    __bf16* h = (__bf16*)(ws + o);  o += (size_t)NTOK * Cc * sizeof(__bf16);     // 16 MiB
    __bf16* W1p = (__bf16*)(ws + o); o += (size_t)Ee * Cc * Hh * sizeof(__bf16); // 64 MiB
    __bf16* W2p = (__bf16*)(ws + o); o += (size_t)Ee * Hh * Cc * sizeof(__bf16); // 64 MiB
    int* counts  = (int*)(ws + o);
    int* counts2 = counts + Ee;
    o += 2 * Ee * sizeof(int);
    int* offsets = (int*)(ws + o);  o += (Ee + 1) * sizeof(int);
    o = (o + 255) & ~(size_t)255;
    int*   tok_ids = (int*)(ws + o);   o += (size_t)NSLOT * sizeof(int);
    float* coefs   = (float*)(ws + o); o += (size_t)NSLOT * sizeof(float);

    // out = x (residual)
    hipMemcpyAsync(out, x, (size_t)NTOK * Cc * sizeof(float),
                   hipMemcpyDeviceToDevice, stream);

    ln_kernel<<<NTOK, 256, 0, stream>>>(x, gamma, beta, h);

    // repack weights to bf16 fragment-major (one pass, reused ~64x per tile)
    {
        size_t thr1 = (size_t)Ee * (Hh / 16) * (Cc / 32) * 32;  // W1: K=Cc, N=Hh
        pack_kernel<<<(unsigned)(thr1 / 256), 256, 0, stream>>>(W1, W1p, Cc, Hh);
        size_t thr2 = (size_t)Ee * (Cc / 16) * (Hh / 32) * 32;  // W2: K=Hh, N=Cc
        pack_kernel<<<(unsigned)(thr2 / 256), 256, 0, stream>>>(W2, W2p, Hh, Cc);
    }

    zero_kernel<<<1, 32, 0, stream>>>(counts);
    count_kernel<<<NSLOT / 256, 256, 0, stream>>>(winners, counts);
    scan_kernel<<<1, 32, 0, stream>>>(counts, offsets);
    fill_kernel<<<NSLOT / 256, 256, 0, stream>>>(winners, weights, offsets,
                                                 counts2, tok_ids, coefs);

    const unsigned shmem = (TM * Cc + TM * NCHUNK) * sizeof(__bf16);  // 80 KB
    moe_kernel<<<dim3(MAXTIL, Ee), 512, shmem, stream>>>(h, W1p, W2p, offsets,
                                                         tok_ids, coefs, out);
}
